// MultiHeadSelfAttention_4612794876257
// MI455X (gfx1250) — matrix operand order repaired and checked by reading, not compiled, not run on hardware
//
#include <hip/hip_runtime.h>

// ---------------------------------------------------------------------------
// Types for CDNA5 WMMA (wave32): v_wmma_f32_16x16x32_bf16
// ---------------------------------------------------------------------------
typedef __attribute__((ext_vector_type(16))) __bf16 v16bf;
typedef __attribute__((ext_vector_type(8)))  __bf16 v8bf;
typedef __attribute__((ext_vector_type(8)))  float  v8f;

// Problem constants
#define BATCH 2
#define TSEQ  2048
#define DMODEL 1024
#define NHEADS 16
#define DHEAD 64
#define E3D   3072           // 3*DMODEL
#define MROWS (BATCH*TSEQ)   // 4096

__device__ __forceinline__ __bf16 f2bf(float f) {
  union { float f; unsigned u; } x; x.f = f;
  unsigned r = x.u + 0x7FFFu + ((x.u >> 16) & 1u);  // round-to-nearest-even
  unsigned short h = (unsigned short)(r >> 16);
  return __builtin_bit_cast(__bf16, h);
}

__device__ __forceinline__ float bf2f(__bf16 b) {
  unsigned short h = __builtin_bit_cast(unsigned short, b);
  unsigned u = ((unsigned)h) << 16;
  return __builtin_bit_cast(float, u);
}

__device__ __forceinline__ v8f wmma_bf16(v16bf a, v16bf b, v8f c) {
  return __builtin_amdgcn_wmma_f32_16x16x32_bf16(
      /*neg_a=*/false, a, /*neg_b=*/false, b,
      /*c_mod=*/(short)0, c, /*reuse_a=*/false, /*reuse_b=*/false);
}

// A-fragment (16x32 MxK, bf16), contiguous layout:
// lane<16: row=lane, elems 0..15 = K[0..16)
// lane>=16: row=lane-16, elems 0..15 = K[16..32)
__device__ __forceinline__ v16bf load_a_frag(const __bf16* __restrict__ A,
                                             int lda, int row0, int k0, int lane) {
  int r    = row0 + (lane & 15);
  int koff = k0 + ((lane >> 4) << 4);
  const __bf16* p = A + (size_t)r * lda + koff;
  union { v16bf v; v8bf h[2]; } u;
  u.h[0] = *(const v8bf*)(p);
  u.h[1] = *(const v8bf*)(p + 8);
  return u.v;
}

// B-fragment (32x16 KxN, bf16) for an NT GEMM: B[k][n] = W[n][k], W row-major
// element i: K = (lane/16)*16 + i, N = lane%16  -> 16 contiguous bf16 of row n
__device__ __forceinline__ v16bf load_b_frag_nt(const __bf16* __restrict__ W,
                                                int ldw, int n0, int k0, int lane) {
  int n    = n0 + (lane & 15);
  int koff = k0 + ((lane >> 4) << 4);
  return *(const v16bf*)(W + (size_t)n * ldw + koff);
}

// ---------------------------------------------------------------------------
// Kernel: elementwise f32 -> bf16 cast (weights)
// ---------------------------------------------------------------------------
__global__ void cast_bf16_kernel(const float* __restrict__ in,
                                 __bf16* __restrict__ out, int n) {
  int i = blockIdx.x * 256 + threadIdx.x;
  if (i < n) out[i] = f2bf(in[i]);
}

// ---------------------------------------------------------------------------
// Kernel: RMSNorm + cast to bf16. One 256-thread block per row.
// ---------------------------------------------------------------------------
__global__ __launch_bounds__(256)
void rmsnorm_cast_kernel(const float* __restrict__ x,
                         const float* __restrict__ w,
                         __bf16* __restrict__ xn) {
  const int D = DMODEL;
  int row = blockIdx.x, tid = threadIdx.x;
  const float* xr = x + (size_t)row * D;
  float ss = 0.f;
  for (int i = tid; i < D; i += 256) { float v = xr[i]; ss += v * v; }
  __shared__ float red[256];
  red[tid] = ss;
  __syncthreads();
  for (int s = 128; s > 0; s >>= 1) {
    if (tid < s) red[tid] += red[tid + s];
    __syncthreads();
  }
  float inv = rsqrtf(red[0] * (1.0f / D) + 1e-6f);
  __bf16* xo = xn + (size_t)row * D;
  for (int i = tid; i < D; i += 256) xo[i] = f2bf(xr[i] * inv * w[i]);
}

// ---------------------------------------------------------------------------
// Kernel: C(MxN) = A(MxK) x W(NxK)^T, bf16 in, f32 accumulate.
// Block: 256 threads = 8 waves (4 M x 2 N); wave tile 32x64; block tile
// 128x128. Ping-pong double-buffered K-loop (no register copies).
// Requires K an odd multiple of 32.
// ---------------------------------------------------------------------------
template <bool OUT_BF16>
__global__ __launch_bounds__(256)
void gemm_nt_kernel(const __bf16* __restrict__ A, const __bf16* __restrict__ W,
                    void* __restrict__ Cv, int M, int N, int K) {
  const int lane = threadIdx.x & 31;
  const int wave = threadIdx.x >> 5;
  const int mbase = blockIdx.x * 128 + (wave >> 1) * 32;  // 4 M-waves
  const int nbase = blockIdx.y * 128 + (wave & 1) * 64;   // 2 N-waves

  v8f acc[2][4] = {};
  v16bf a0[2], b0[4], a1[2], b1[4];

  auto load_set = [&](v16bf* aa, v16bf* bb, int kb) {
    aa[0] = load_a_frag(A, K, mbase,      kb, lane);
    aa[1] = load_a_frag(A, K, mbase + 16, kb, lane);
#pragma unroll
    for (int j = 0; j < 4; j++)
      bb[j] = load_b_frag_nt(W, K, nbase + j * 16, kb, lane);
  };
  auto compute_set = [&](const v16bf* aa, const v16bf* bb) {
#pragma unroll
    for (int i = 0; i < 2; i++)
#pragma unroll
      for (int j = 0; j < 4; j++)
        acc[i][j] = wmma_bf16(aa[i], bb[j], acc[i][j]);
  };

  load_set(a0, b0, 0);
  for (int kb = 32; kb + 32 < K; kb += 64) {
    load_set(a1, b1, kb);        // prefetch block kb
    compute_set(a0, b0);         // compute block kb-32
    load_set(a0, b0, kb + 32);   // prefetch block kb+32
    compute_set(a1, b1);         // compute block kb
  }
  compute_set(a0, b0);           // final block (K-32)

  const int cn = lane & 15, hg = lane >> 4;
#pragma unroll
  for (int i = 0; i < 2; i++)
#pragma unroll
    for (int j = 0; j < 4; j++)
#pragma unroll
      for (int r = 0; r < 8; r++) {
        int row = mbase + i * 16 + hg * 8 + r;
        int col = nbase + j * 16 + cn;
        if constexpr (OUT_BF16)
          ((__bf16*)Cv)[(size_t)row * N + col] = f2bf(acc[i][j][r]);
        else
          ((float*)Cv)[(size_t)row * N + col] = acc[i][j][r];
      }
}

// ---------------------------------------------------------------------------
// Kernel: causal flash-attention, TRANSPOSED formulation.
// Computes S^T = K x Q^T per key tile, so queries live in lanes and keys in
// registers: softmax stats are per-lane scalars (1 shfl_xor per stat), the
// P^T B-fragment for ctx^T = V^T x P^T needs only one cross-half register
// swap (no LDS round trip), and the epilogue stores 16B per fragment.
// 128 threads = 4 waves; each wave owns a 16-query tile, key blocks of 32.
// qkv layout: [b*T + t][3072] with q @ +0, k @ +1024, v @ +2048 (+h*64).
// ---------------------------------------------------------------------------
__global__ __launch_bounds__(128)
void attn_kernel(const __bf16* __restrict__ qkv, __bf16* __restrict__ ctx) {
  const int lane = threadIdx.x & 31;
  const int wave = threadIdx.x >> 5;
  const int ln = lane & 15, hg = lane >> 4;
  const int b = blockIdx.y >> 4, h = blockIdx.y & 15;
  const int qb = blockIdx.x * 64 + wave * 16;
  const int bT = b * TSEQ;

  __shared__ alignas(32) __bf16 VT[4][64 * 40];    // V^T tile [d][key]

  // Q as B-operand (Q^T: B[k=d][n=query]), loaded once, scale 1/sqrt(64) folded
  const __bf16* Qp = qkv + (size_t)(bT + qb) * E3D + h * DHEAD;
  v16bf qf0 = load_b_frag_nt(Qp, E3D, 0, 0, lane);
  v16bf qf1 = load_b_frag_nt(Qp, E3D, 0, 32, lane);
#pragma unroll
  for (int e = 0; e < 16; e++) {
    qf0[e] = f2bf(bf2f(qf0[e]) * 0.125f);
    qf1[e] = f2bf(bf2f(qf1[e]) * 0.125f);
  }

  // per-lane softmax state: this lane's query is qb + ln (replicated over hg)
  float m = -1e30f, lsum = 0.f;
  v8f acc[4] = {};   // ctx^T C-layout: acc[n][r] <-> d = n*16 + hg*8 + r, q = ln

  // per-lane V staging assignment: key octet + d chunk
  const int vko = (lane >> 3) * 8;     // key octet base: 0,8,16,24
  const int vd0 = (lane & 7) * 8;      // d chunk base: 0..56

  const int jend = qb + 16;
  // last fully-causal (mask-free) block boundary: jb+31 <= qb
  const int jfull = (qb >= 31) ? ((((qb - 31) >> 5) + 1) << 5) : 0;

  auto process_block = [&](int jb, bool masked) {
    // ---- prefetch next block's K/V rows (global_prefetch_b8)
    if (jb + 32 < jend && jb + 32 + lane < TSEQ) {
      const __bf16* nrow = qkv + (size_t)(bT + jb + 32 + lane) * E3D + h * DHEAD;
      __builtin_prefetch(nrow + DMODEL, 0, 3);      // next K rows
      __builtin_prefetch(nrow + 2 * DMODEL, 0, 3);  // next V rows
    }

    // ---- stage V^T (32 keys x 64 dims) into LDS: VT[d][key]
    // per-lane 8x8 tile: 8 coalesced b128 loads, register transpose,
    // 8 ds_store_b128 (16B each)
    const __bf16* Vb = qkv + (size_t)(bT + jb) * E3D + 2 * DMODEL + h * DHEAD;
    v8bf vin[8];
#pragma unroll
    for (int j = 0; j < 8; j++)
      vin[j] = *(const v8bf*)(Vb + (size_t)(vko + j) * E3D + vd0);
#pragma unroll
    for (int d = 0; d < 8; d++) {
      v8bf t;
#pragma unroll
      for (int j = 0; j < 8; j++) t[j] = vin[j][d];
      *(v8bf*)(&VT[wave][(vd0 + d) * 40 + vko]) = t;
    }

    // ---- scores S^T = K x Q^T for two 16-key tiles
    // pv[t*8+r] <-> key_local = t*16 + hg*8 + r, query = qb + ln
    float pv[16];
#pragma unroll
    for (int t = 0; t < 2; t++) {
      const __bf16* Kb =
          qkv + (size_t)(bT + jb + t * 16) * E3D + DMODEL + h * DHEAD;
      v16bf ka0 = load_a_frag(Kb, E3D, 0, 0, lane);
      v16bf ka1 = load_a_frag(Kb, E3D, 0, 32, lane);
      v8f s = {};
      s = wmma_bf16(ka0, qf0, s);
      s = wmma_bf16(ka1, qf1, s);
#pragma unroll
      for (int r = 0; r < 8; r++) pv[t * 8 + r] = s[r];
    }

    // ---- causal mask (only for diagonal blocks)
    if (masked) {
      int query = qb + ln;
#pragma unroll
      for (int t = 0; t < 2; t++)
#pragma unroll
        for (int r = 0; r < 8; r++) {
          int key = jb + t * 16 + hg * 8 + r;
          if (key > query) pv[t * 8 + r] = -1e30f;
        }
    }

    // ---- online softmax: in-lane tree + single cross-half shuffle per stat
    float rm = pv[0];
#pragma unroll
    for (int i = 1; i < 16; i++) rm = fmaxf(rm, pv[i]);
    rm = fmaxf(rm, __shfl_xor(rm, 16, 32));
    float mn = fmaxf(m, rm);
    float alpha = __expf(m - mn);
    float rs = 0.f;
#pragma unroll
    for (int i = 0; i < 16; i++) { pv[i] = __expf(pv[i] - mn); rs += pv[i]; }
    rs += __shfl_xor(rs, 16, 32);
    lsum = lsum * alpha + rs;
    m = mn;
#pragma unroll
    for (int n = 0; n < 4; n++)
#pragma unroll
      for (int r = 0; r < 8; r++) acc[n][r] *= alpha;

    // ---- build P^T B-fragment: b[i] = P^T[key = hg*16+i][q = ln]
    // present per lane: hg=0 has keys 0-7 (t0), 16-23 (t1);
    //                   hg=1 has keys 8-15 (t0), 24-31 (t1).
    // one xor-16 swap moves the missing 8 keys into each half.
    float recv[8];
#pragma unroll
    for (int r = 0; r < 8; r++)
      recv[r] = __shfl_xor(hg ? pv[r] : pv[8 + r], 16, 32);
    v16bf pb;
#pragma unroll
    for (int i = 0; i < 8; i++) {
      pb[i]     = f2bf(hg ? recv[i] : pv[i]);
      pb[8 + i] = f2bf(hg ? pv[8 + i] : recv[i]);
    }

    // ---- ctx^T += V^T @ P^T  (A-frag = V^T rows from LDS)
    asm volatile("s_wait_dscnt 0" ::: "memory");  // VT stores visible
#pragma unroll
    for (int n = 0; n < 4; n++) {
      union { v16bf v; v8bf hx[2]; } va;
      const __bf16* p = &VT[wave][(n * 16 + ln) * 40 + hg * 16];
      va.hx[0] = *(const v8bf*)(p);
      va.hx[1] = *(const v8bf*)(p + 8);
      acc[n] = wmma_bf16(va.v, pb, acc[n]);
    }
  };

  for (int jb = 0; jb < jfull; jb += 32) process_block(jb, false);
  for (int jb = jfull; jb < jend; jb += 32) process_block(jb, true);

  // ---- normalize and store ctx: lane (ln,hg) owns query qb+ln,
  // d = n*16 + hg*8 + r  -> 8 contiguous bf16 per fragment = 16B stores
  float rinv = 1.0f / lsum;
  __bf16* orow = ctx + (size_t)(bT + qb + ln) * DMODEL + h * DHEAD;
#pragma unroll
  for (int n = 0; n < 4; n++) {
    v8bf o;
#pragma unroll
    for (int r = 0; r < 8; r++) o[r] = f2bf(acc[n][r] * rinv);
    *(v8bf*)(orow + n * 16 + hg * 8) = o;
  }
}

// ---------------------------------------------------------------------------
// Host-side launcher
// ---------------------------------------------------------------------------
extern "C" void kernel_launch(void* const* d_in, const int* in_sizes, int n_in,
                              void* d_out, int out_size, void* d_ws, size_t ws_size,
                              hipStream_t stream) {
  (void)in_sizes; (void)n_in; (void)out_size; (void)ws_size;
  const float* x      = (const float*)d_in[0];  // (2,2048,1024)
  const float* nw     = (const float*)d_in[1];  // (1024,)
  const float* qkvw_f = (const float*)d_in[2];  // (3072,1024)
  const float* outw_f = (const float*)d_in[3];  // (1024,1024)
  float* out = (float*)d_out;                   // (2,2048,1024)

  char* ws = (char*)d_ws;
  __bf16* xn   = (__bf16*)(ws);                       //  8 MB: normed x
  __bf16* qkvw = (__bf16*)(ws + (8u  << 20));         //  6 MB: qkv_w bf16
  __bf16* outw = (__bf16*)(ws + (14u << 20));         //  2 MB: out_w bf16
  __bf16* qkv  = (__bf16*)(ws + (16u << 20));         // 24 MB: qkv activations
  __bf16* ctx  = (__bf16*)(ws + (40u << 20));         //  8 MB: attention output

  cast_bf16_kernel<<<(E3D * DMODEL + 255) / 256, 256, 0, stream>>>(
      qkvw_f, qkvw, E3D * DMODEL);
  cast_bf16_kernel<<<(DMODEL * DMODEL + 255) / 256, 256, 0, stream>>>(
      outw_f, outw, DMODEL * DMODEL);

  rmsnorm_cast_kernel<<<MROWS, 256, 0, stream>>>(x, nw, xn);

  gemm_nt_kernel<true><<<dim3(MROWS / 128, E3D / 128), 256, 0, stream>>>(
      xn, qkvw, qkv, MROWS, E3D, DMODEL);

  attn_kernel<<<dim3(TSEQ / 64, BATCH * NHEADS), 128, 0, stream>>>(qkv, ctx);

  gemm_nt_kernel<false><<<dim3(MROWS / 128, DMODEL / 128), 256, 0, stream>>>(
      ctx, outw, out, MROWS, DMODEL, DMODEL);
}
